// GlobalAttentionClassifier_77704548319506
// MI455X (gfx1250) — compile-verified, matrix operand order checked by reading, not run on hardware
//
#include <hip/hip_runtime.h>
#include <hip/hip_bf16.h>
#include <stdint.h>

typedef __attribute__((ext_vector_type(16))) _Float16 v16h;
typedef __attribute__((ext_vector_type(8)))  float    v8f;

#define HDIM 128
#define OUTD 10

// ---------------- utility kernels ----------------

__global__ void k_fill_f32(float* p, float v, int n) {
    int i = blockIdx.x * blockDim.x + threadIdx.x;
    if (i < n) p[i] = v;
}

__global__ void k_count_deg(const int* __restrict__ col, float* deg, int e) {
    int i = blockIdx.x * blockDim.x + threadIdx.x;
    if (i < e) atomicAdd(&deg[col[i]], 1.0f);
}

__global__ void k_deg_to_dis(float* deg, int n) {
    int i = blockIdx.x * blockDim.x + threadIdx.x;
    if (i < n) deg[i] = rsqrtf(deg[i]);   // deg >= 1 always (self loop)
}

// Pack a 128x128 f32 weight (row-major [k][n]) into the exact WMMA B-operand
// register image, as f16:  P[((tile*4 + kc)*32 + lane)*16 + e]
//   lane: ml = lane&15 (N within tile), hi = lane>>4 (K half select)
//   e = 2v (+1): K = kc*32 + hi*8 + (v<4 ? 2v : 16+2(v-4))
// 1024 threads per weight matrix; each thread emits one 32B lane record.
__global__ void k_pack_w(const float* __restrict__ W, _Float16* __restrict__ P) {
    int t = blockIdx.x * blockDim.x + threadIdx.x;
    if (t >= 8 * 4 * 32) return;
    const int lane = t & 31;
    const int kc   = (t >> 5) & 3;
    const int tile = t >> 7;
    const int ml = lane & 15;
    const int hi = lane >> 4;
    _Float16* dst = P + (size_t)t * 16;
    const int kb = kc * 32 + hi * 8;
    #pragma unroll
    for (int v = 0; v < 8; ++v) {
        const int k0 = kb + ((v < 4) ? 2 * v : 16 + 2 * (v - 4));
        dst[2 * v]     = (_Float16)W[k0 * HDIM + tile * 16 + ml];
        dst[2 * v + 1] = (_Float16)W[(k0 + 1) * HDIM + tile * 16 + ml];
    }
}

// ---------------- WMMA GEMM: out[M x 128] = in[M x 128] @ W[128 x 128] ----------------
// block = 256 threads (8 waves); wave w owns col tile [16w,16w+16); block owns 64 rows.
// B operands pre-packed in register-image order: one 32B vector load per k-chunk,
// held in VGPRs and reused across 4 M-tiles (16 WMMAs per wave).
__global__ void k_gemm_128(const float* __restrict__ X, const v16h* __restrict__ Wp,
                           float* __restrict__ out, const float* __restrict__ bias,
                           int nrows, int do_relu) {
    const int lane = threadIdx.x & 31;
    const int wave = threadIdx.x >> 5;        // 0..7 -> N tile
    const int row0 = blockIdx.x * 64;
    const int col0 = wave * 16;
    const int ml   = lane & 15;
    const int hi   = lane >> 4;               // half-wave K select

    // load this wave's 4 B k-slices (coalesced b128 pairs), reused 4x
    v16h b[4];
    #pragma unroll
    for (int kc = 0; kc < 4; ++kc)
        b[kc] = Wp[(size_t)(wave * 4 + kc) * 32 + lane];

    v8f acc[4];
    #pragma unroll
    for (int mt = 0; mt < 4; ++mt) acc[mt] = (v8f){};

    #pragma unroll
    for (int mt = 0; mt < 4; ++mt) {
        int row = row0 + mt * 16 + ml;
        if (row >= nrows) row = nrows - 1;    // branchless clamp; EXEC stays full
        const float* xr = X + (long)row * HDIM;
        #pragma unroll
        for (int kc = 0; kc < 4; ++kc) {      // K = 4 x 32
            const int kb = kc * 32 + hi * 8;
            v16h a;
            #pragma unroll
            for (int j = 0; j < 8; ++j) {     // contiguous: vectorizes to b128 + cvt_pk
                a[j]     = (_Float16)xr[kb + j];
                a[8 + j] = (_Float16)xr[kb + 16 + j];
            }
            acc[mt] = __builtin_amdgcn_wmma_f32_16x16x32_f16(false, a, false, b[kc],
                                                             (short)0, acc[mt], false, false);
        }
    }

    const float bv = bias ? bias[col0 + ml] : 0.0f;
    #pragma unroll
    for (int mt = 0; mt < 4; ++mt) {
        #pragma unroll
        for (int v = 0; v < 8; ++v) {
            const int m = v + 8 * hi;         // C/D layout: M = vgpr + 8*half
            const int r = row0 + mt * 16 + m;
            if (r < nrows) {
                float val = acc[mt][v] + bv;
                if (do_relu) val = fmaxf(val, 0.0f);
                out[(long)r * HDIM + col0 + ml] = val;
            }
        }
    }
}

// ---------------- aggregation ----------------

// agg[i,:] = dis[i]^2 * h[i,:]   (self loop term)
__global__ void k_self_init(const float* __restrict__ h, const float* __restrict__ dis,
                            float* __restrict__ agg, int n) {
    long i = (long)blockIdx.x * blockDim.x + threadIdx.x;
    if (i < (long)n * HDIM) {
        float d = dis[i >> 7];
        agg[i] = d * d * h[i];
    }
}

// one wave32 per edge; lane moves 4 consecutive floats (coalesced 512B row read)
__global__ void k_edge_agg(const int* __restrict__ row, const int* __restrict__ col,
                           const float* __restrict__ dis, const float* __restrict__ h,
                           float* __restrict__ agg, int e) {
    int wid  = (blockIdx.x * blockDim.x + threadIdx.x) >> 5;
    int lane = threadIdx.x & 31;
    if (wid >= e) return;
    int r = row[wid], c = col[wid];
    float w = dis[r] * dis[c];
    float4 v = ((const float4*)(h + (long)r * HDIM))[lane];
    float* a = agg + (long)c * HDIM + lane * 4;
    atomicAdd(a + 0, w * v.x);
    atomicAdd(a + 1, w * v.y);
    atomicAdd(a + 2, w * v.z);
    atomicAdd(a + 3, w * v.w);
}

__global__ void k_bias_relu(float* __restrict__ a, const float* __restrict__ bias, int n) {
    long i = (long)blockIdx.x * blockDim.x + threadIdx.x;
    if (i < (long)n * HDIM) a[i] = fmaxf(a[i] + bias[i & (HDIM - 1)], 0.0f);
}

// ---------------- attention pooling ----------------

// gate[i] = dot(g1[i,:], gW2) + gb2 ; one wave per node, shfl reduction
__global__ void k_gate_dot(const float* __restrict__ g1, const float* __restrict__ gW2,
                           const float* __restrict__ gb2, float* __restrict__ gate, int n) {
    int wid  = (blockIdx.x * blockDim.x + threadIdx.x) >> 5;
    int lane = threadIdx.x & 31;
    if (wid >= n) return;
    float4 x = ((const float4*)(g1 + (long)wid * HDIM))[lane];
    float4 w = ((const float4*)gW2)[lane];
    float s = x.x * w.x + x.y * w.y + x.z * w.z + x.w * w.w;
    #pragma unroll
    for (int o = 16; o; o >>= 1) s += __shfl_xor(s, o, 32);
    if (lane == 0) gate[wid] = s + gb2[0];
}

__device__ inline void atomicMaxFloat(float* addr, float val) {
    if (val >= 0.0f)
        atomicMax((int*)addr, __float_as_int(val));
    else
        atomicMin((unsigned int*)addr, __float_as_uint(val));
}

__global__ void k_seg_max(const float* __restrict__ gate, const int* __restrict__ batch,
                          float* gmax, int n) {
    int i = blockIdx.x * blockDim.x + threadIdx.x;
    if (i < n) atomicMaxFloat(&gmax[batch[i]], gate[i]);
}

__global__ void k_seg_exp(const float* __restrict__ gate, const int* __restrict__ batch,
                          const float* __restrict__ gmax, float* __restrict__ att,
                          float* denom, int n) {
    int i = blockIdx.x * blockDim.x + threadIdx.x;
    if (i < n) {
        float e = expf(gate[i] - gmax[batch[i]]);
        att[i] = e;
        atomicAdd(&denom[batch[i]], e);
    }
}

// pooled[b,:] += (att[i]/denom[b]) * h[i,:] ; one wave per node
__global__ void k_pool(const float* __restrict__ att, const float* __restrict__ denom,
                       const int* __restrict__ batch, const float* __restrict__ h,
                       float* __restrict__ pooled, int n) {
    int wid  = (blockIdx.x * blockDim.x + threadIdx.x) >> 5;
    int lane = threadIdx.x & 31;
    if (wid >= n) return;
    int b = batch[wid];
    float w = att[wid] / denom[b];
    float4 v = ((const float4*)(h + (long)wid * HDIM))[lane];
    float* p = pooled + (long)b * HDIM + lane * 4;
    atomicAdd(p + 0, w * v.x);
    atomicAdd(p + 1, w * v.y);
    atomicAdd(p + 2, w * v.z);
    atomicAdd(p + 3, w * v.w);
}

// ---------------- MLP head (G=512: tiny) ----------------

__global__ void k_head1(const float* __restrict__ pooled, const float* __restrict__ lW1,
                        const float* __restrict__ lb1, float* __restrict__ hidden) {
    __shared__ float p[HDIM];
    int g = blockIdx.x, t = threadIdx.x;
    p[t] = pooled[(long)g * HDIM + t];
    __syncthreads();
    float s = lb1[t];
    #pragma unroll 4
    for (int k = 0; k < HDIM; ++k) s += p[k] * lW1[k * HDIM + t];
    hidden[(long)g * HDIM + t] = fmaxf(s, 0.0f);
}

__global__ void k_head2(const float* __restrict__ hidden, const float* __restrict__ lW2,
                        const float* __restrict__ lb2, float* __restrict__ out,
                        int g_cnt, int out_size) {
    int i = blockIdx.x * blockDim.x + threadIdx.x;
    if (i >= out_size) return;
    if (i >= g_cnt * OUTD) { out[i] = 0.0f; return; }  // aux scalar(s)
    int g = i / OUTD, o = i % OUTD;
    float s = lb2[o];
    #pragma unroll 4
    for (int k = 0; k < HDIM; ++k) s += hidden[(long)g * HDIM + k] * lW2[k * OUTD + o];
    out[i] = s;
}

// ---------------- host ----------------

extern "C" void kernel_launch(void* const* d_in, const int* in_sizes, int n_in,
                              void* d_out, int out_size, void* d_ws, size_t ws_size,
                              hipStream_t stream) {
    const float* x    = (const float*)d_in[0];
    const int*   ei   = (const int*)d_in[1];
    const int*   batch= (const int*)d_in[2];
    const float* W[3] = {(const float*)d_in[3], (const float*)d_in[5], (const float*)d_in[7]};
    const float* B[3] = {(const float*)d_in[4], (const float*)d_in[6], (const float*)d_in[8]};
    const float* gW1  = (const float*)d_in[9];
    const float* gb1  = (const float*)d_in[10];
    const float* gW2  = (const float*)d_in[11];
    const float* gb2  = (const float*)d_in[12];
    const float* lW1  = (const float*)d_in[13];
    const float* lb1  = (const float*)d_in[14];
    const float* lW2  = (const float*)d_in[15];
    const float* lb2  = (const float*)d_in[16];

    const int N = in_sizes[0] / HDIM;
    const int E = in_sizes[1] / 2;
    const int G = (out_size > 1) ? (out_size - 1) / OUTD : 512;
    const int* erow = ei;
    const int* ecol = ei + E;

    // workspace layout (256B aligned)
    char* base = (char*)d_ws;
    size_t off = 0;
    auto alloc = [&](size_t bytes) { void* p = base + off; off = (off + bytes + 255) & ~(size_t)255; return p; };
    float*    dis    = (float*)alloc((size_t)N * 4);
    float*    bufA   = (float*)alloc((size_t)N * HDIM * 4);   // activations / agg
    float*    bufB   = (float*)alloc((size_t)N * HDIM * 4);   // gemm output h / g1
    float*    gate   = (float*)alloc((size_t)N * 4);
    float*    att    = (float*)alloc((size_t)N * 4);
    float*    gmax   = (float*)alloc((size_t)G * 4);
    float*    denom  = (float*)alloc((size_t)G * 4);
    float*    pooled = (float*)alloc((size_t)G * HDIM * 4);
    float*    hidden = (float*)alloc((size_t)G * HDIM * 4);
    _Float16* wf16   = (_Float16*)alloc((size_t)4 * HDIM * HDIM * 2); // packed W0,W1,W2,gW1

    const int T = 256;
    auto cdiv = [](long a, long b) { return (int)((a + b - 1) / b); };

    // 1) gcn_norm: deg (self loop = 1) -> dis = rsqrt(deg)
    k_fill_f32<<<cdiv(N, T), T, 0, stream>>>(dis, 1.0f, N);
    k_count_deg<<<cdiv(E, T), T, 0, stream>>>(ecol, dis, E);
    k_deg_to_dis<<<cdiv(N, T), T, 0, stream>>>(dis, N);

    // 2) pack weights into WMMA B-operand register image (f16)
    const int WSZ = HDIM * HDIM;
    k_pack_w<<<4, T, 0, stream>>>(W[0], wf16 + 0 * WSZ);
    k_pack_w<<<4, T, 0, stream>>>(W[1], wf16 + 1 * WSZ);
    k_pack_w<<<4, T, 0, stream>>>(W[2], wf16 + 2 * WSZ);
    k_pack_w<<<4, T, 0, stream>>>(gW1,  wf16 + 3 * WSZ);

    const int gemm_blocks = cdiv(N, 64);
    const int nh = cdiv((long)N * HDIM, T);
    const int ew = cdiv((long)E * 32, T);   // wave per edge
    const int nw = cdiv((long)N * 32, T);   // wave per node

    // 3) three GCN layers
    for (int l = 0; l < 3; ++l) {
        const float* in = (l == 0) ? x : bufA;
        k_gemm_128<<<gemm_blocks, T, 0, stream>>>(in, (const v16h*)(wf16 + l * WSZ),
                                                  bufB, nullptr, N, 0);
        k_self_init<<<nh, T, 0, stream>>>(bufB, dis, bufA, N);
        k_edge_agg<<<ew, T, 0, stream>>>(erow, ecol, dis, bufB, bufA, E);
        k_bias_relu<<<nh, T, 0, stream>>>(bufA, B[l], N);
    }

    // 4) gate network + segment softmax + pooling
    k_gemm_128<<<gemm_blocks, T, 0, stream>>>(bufA, (const v16h*)(wf16 + 3 * WSZ),
                                              bufB, gb1, N, 1);
    k_gate_dot<<<nw, T, 0, stream>>>(bufB, gW2, gb2, gate, N);

    k_fill_f32<<<cdiv(G, T), T, 0, stream>>>(gmax, -__builtin_huge_valf(), G);
    k_fill_f32<<<cdiv(G, T), T, 0, stream>>>(denom, 0.0f, G);
    k_fill_f32<<<cdiv((long)G * HDIM, T), T, 0, stream>>>(pooled, 0.0f, G * HDIM);

    k_seg_max<<<cdiv(N, T), T, 0, stream>>>(gate, batch, gmax, N);
    k_seg_exp<<<cdiv(N, T), T, 0, stream>>>(gate, batch, gmax, att, denom, N);
    k_pool<<<nw, T, 0, stream>>>(att, denom, batch, bufA, pooled, N);

    // 5) MLP head + aux scalar
    k_head1<<<G, HDIM, 0, stream>>>(pooled, lW1, lb1, hidden);
    k_head2<<<cdiv(out_size, T), T, 0, stream>>>(hidden, lW2, lb2, (float*)d_out, G, out_size);
}
